// EfficientAttention_47236050321895
// MI455X (gfx1250) — matrix operand order repaired and checked by reading, not compiled, not run on hardware
//
#include <hip/hip_runtime.h>
#include <hip/hip_bf16.h>

#define DMODEL 256
#define NHEAD 8
#define HDIM 32

typedef __bf16 v16bf __attribute__((ext_vector_type(16)));
typedef __bf16 v8bf  __attribute__((ext_vector_type(8)));
typedef float  v8f   __attribute__((ext_vector_type(8)));

__device__ __forceinline__ __bf16 f2bf(float f) {
  union { float f; unsigned u; } v; v.f = f;
  unsigned r = (v.u + 0x7FFFu + ((v.u >> 16) & 1u)) >> 16;  // round-to-nearest-even
  union { unsigned short s; __bf16 b; } o; o.s = (unsigned short)r;
  return o.b;
}

// ---------------- elementwise f32 -> bf16 ----------------
__global__ void cvt_f32_bf16(const float* __restrict__ src, __bf16* __restrict__ dst, int n) {
  int i = blockIdx.x * blockDim.x + threadIdx.x;
  if (i < n) dst[i] = f2bf(src[i]);
}

// ---------------- C[M,256] = A[M,256] * B[256,256]^T (+bias), bf16 WMMA ----------------
// Block = 256 threads = 8 waves; block tile = 32(M) x 256(N); grid.x = M/32.
// Each wave owns a 32(M) x 32(N) tile: 2 A-fragments x 2 B-fragments -> 4 accumulators,
// 32 WMMAs per wave, 2 b128 loads per WMMA (vs 3 with a 16x32 tile).
// A fragment (16x32 bf16): lane<16 holds row=lane%16, K={kb..kb+7, kb+16..kb+23};
//                          lane>=16 holds K={kb+8..kb+15, kb+24..kb+31}.
// B fragment (32x16 bf16): lane holds col=lane%16, same K split as A fragment.
__global__ __launch_bounds__(256) void gemm_xwt(const __bf16* __restrict__ A,
                                                const __bf16* __restrict__ B,
                                                const float* __restrict__ bias,
                                                float* __restrict__ C) {
  const int lane   = threadIdx.x & 31;
  const int wave   = threadIdx.x >> 5;
  const int m0     = blockIdx.x * 32;
  const int n0     = wave * 32;            // 8 waves * 32 cols = 256 = full N
  const int laneHi = lane >> 4;            // 0 or 1
  const int laneLo = lane & 15;

  const __bf16* arow0 = A + (size_t)(m0 + laneLo) * DMODEL;
  const __bf16* arow1 = arow0 + (size_t)16 * DMODEL;
  const __bf16* bcol0 = B + (size_t)(n0 + laneLo) * DMODEL;
  const __bf16* bcol1 = B + (size_t)(n0 + 16 + laneLo) * DMODEL;

  v8f acc00 = {}; v8f acc01 = {}; v8f acc10 = {}; v8f acc11 = {};
  union Frag { v16bf v; v8bf h[2]; };

#pragma unroll
  for (int kb = 0; kb < DMODEL; kb += 32) {
    if (kb + 32 < DMODEL) {                // CDNA5 speculative prefetch of next K-slab
      __builtin_prefetch(arow0 + kb + 32, 0, 1);
      __builtin_prefetch(arow1 + kb + 32, 0, 1);
      __builtin_prefetch(bcol0 + kb + 32, 0, 1);
    }
    Frag a0, a1, b0, b1;
    a0.h[0] = *(const v8bf*)(arow0 + kb + laneHi * 8);
    a0.h[1] = *(const v8bf*)(arow0 + kb + laneHi * 8 + 16);
    a1.h[0] = *(const v8bf*)(arow1 + kb + laneHi * 8);
    a1.h[1] = *(const v8bf*)(arow1 + kb + laneHi * 8 + 16);
    b0.h[0] = *(const v8bf*)(bcol0 + kb + laneHi * 8);
    b0.h[1] = *(const v8bf*)(bcol0 + kb + laneHi * 8 + 16);
    b1.h[0] = *(const v8bf*)(bcol1 + kb + laneHi * 8);
    b1.h[1] = *(const v8bf*)(bcol1 + kb + laneHi * 8 + 16);
    acc00 = __builtin_amdgcn_wmma_f32_16x16x32_bf16(false, a0.v, false, b0.v,
                                                    (short)0, acc00, false, false);
    acc01 = __builtin_amdgcn_wmma_f32_16x16x32_bf16(false, a0.v, false, b1.v,
                                                    (short)0, acc01, false, false);
    acc10 = __builtin_amdgcn_wmma_f32_16x16x32_bf16(false, a1.v, false, b0.v,
                                                    (short)0, acc10, false, false);
    acc11 = __builtin_amdgcn_wmma_f32_16x16x32_bf16(false, a1.v, false, b1.v,
                                                    (short)0, acc11, false, false);
  }

  const int rowAdd = laneHi * 8;           // C/D layout: VGPR r -> M=r (lanes 0-15), M=r+8 (lanes 16-31)
  const int c0 = n0 + laneLo;
  const int c1 = n0 + 16 + laneLo;
  const float bb0 = bias ? bias[c0] : 0.0f;
  const float bb1 = bias ? bias[c1] : 0.0f;
#pragma unroll
  for (int r = 0; r < 8; ++r) {
    const int rowA = m0 + r + rowAdd;       // acc00 / acc01
    const int rowB = rowA + 16;             // acc10 / acc11
    C[(size_t)rowA * DMODEL + c0] = acc00[r] + bb0;
    C[(size_t)rowA * DMODEL + c1] = acc01[r] + bb1;
    C[(size_t)rowB * DMODEL + c0] = acc10[r] + bb0;
    C[(size_t)rowB * DMODEL + c1] = acc11[r] + bb1;
  }
}

// ---------------- column sums of V: vsum[d] = sum_n V[n,d] ----------------
__global__ void col_sum(const float* __restrict__ V, float* __restrict__ vsum, int Nrows) {
  __shared__ float sm[256];
  const int col = blockIdx.x;
  float s = 0.f;
  for (int r = threadIdx.x; r < Nrows; r += 256) s += V[(size_t)r * DMODEL + col];
  sm[threadIdx.x] = s;
  __syncthreads();
  for (int off = 128; off > 0; off >>= 1) {
    if ((int)threadIdx.x < off) sm[threadIdx.x] += sm[threadIdx.x + off];
    __syncthreads();
  }
  if (threadIdx.x == 0) vsum[col] = sm[0];
}

// ---------------- num[n,:] = vsum ; denom[n,h] = N ----------------
__global__ void init_num_denom(const float* __restrict__ vsum, float* __restrict__ num,
                               float* __restrict__ denom, int Nrows) {
  const int n = blockIdx.x, d = threadIdx.x;
  num[(size_t)n * DMODEL + d] = vsum[d];
  if (d < NHEAD) denom[n * NHEAD + d] = (float)Nrows;
}

// ---------------- one wave32 per (edge, head) ----------------
// w = exp(<Q[row,h],K[col,h]>/sqrt(HD)) - 1 ; num[row,h,:] += w*V[col,h,:]; denom[row,h] += w
__global__ __launch_bounds__(256) void edge_scatter(const int* __restrict__ ei, int E,
                                                    const float* __restrict__ Q,
                                                    const float* __restrict__ K,
                                                    const float* __restrict__ V,
                                                    float* __restrict__ num,
                                                    float* __restrict__ denom) {
  const int wid  = blockIdx.x * 8 + (threadIdx.x >> 5);
  const int lane = threadIdx.x & 31;
  const int e = wid >> 3;     // / NHEAD
  const int h = wid & (NHEAD - 1);
  if (e >= E) return;
  const int row = ei[e];
  const int col = ei[E + e];
  const size_t qoff = (size_t)row * DMODEL + h * HDIM + lane;
  const size_t koff = (size_t)col * DMODEL + h * HDIM + lane;
  float p = Q[qoff] * K[koff];
#pragma unroll
  for (int off = 16; off > 0; off >>= 1) p += __shfl_xor(p, off, 32);  // wave32 reduction
  const float w = __expf(p * 0.17677669529663687f) - 1.0f;             // 1/sqrt(32)
  atomicAdd(&num[qoff], w * V[koff]);
  if (lane == 0) atomicAdd(&denom[row * NHEAD + h], w);
}

// ---------------- pre_b[n,d] = bf16(num[n,d] / denom[n, d/32]) ----------------
__global__ void norm_cvt(const float* __restrict__ num, const float* __restrict__ denom,
                         __bf16* __restrict__ pre) {
  const int n = blockIdx.x, d = threadIdx.x;
  const size_t idx = (size_t)n * DMODEL + d;
  pre[idx] = f2bf(num[idx] / denom[n * NHEAD + (d >> 5)]);
}

extern "C" void kernel_launch(void* const* d_in, const int* in_sizes, int n_in,
                              void* d_out, int out_size, void* d_ws, size_t ws_size,
                              hipStream_t stream) {
  const float* x  = (const float*)d_in[0];
  const int*   ei = (const int*)d_in[1];
  const float* Wq = (const float*)d_in[2];
  const float* Wk = (const float*)d_in[3];
  const float* Wv = (const float*)d_in[4];
  const float* Wo = (const float*)d_in[5];
  const float* bo = (const float*)d_in[6];
  float* out = (float*)d_out;

  const int N = in_sizes[0] / DMODEL;
  const int E = in_sizes[1] / 2;
  const int WW = DMODEL * DMODEL;

  // workspace carve-out (~21 MB)
  char* ws = (char*)d_ws;
  size_t off = 0;
  auto carve = [&](size_t bytes) -> void* {
    void* p = ws + off;
    off = (off + bytes + 255) & ~(size_t)255;
    return p;
  };
  __bf16* xb   = (__bf16*)carve((size_t)N * DMODEL * sizeof(__bf16));
  __bf16* wqb  = (__bf16*)carve((size_t)WW * sizeof(__bf16));
  __bf16* wkb  = (__bf16*)carve((size_t)WW * sizeof(__bf16));
  __bf16* wvb  = (__bf16*)carve((size_t)WW * sizeof(__bf16));
  __bf16* wob  = (__bf16*)carve((size_t)WW * sizeof(__bf16));
  float*  Qf   = (float*)carve((size_t)N * DMODEL * sizeof(float));
  float*  Kf   = (float*)carve((size_t)N * DMODEL * sizeof(float));
  float*  Vf   = (float*)carve((size_t)N * DMODEL * sizeof(float));
  float*  num  = (float*)carve((size_t)N * DMODEL * sizeof(float));
  float*  den  = (float*)carve((size_t)N * NHEAD * sizeof(float));
  float*  vsum = (float*)carve((size_t)DMODEL * sizeof(float));
  __bf16* preb = (__bf16*)carve((size_t)N * DMODEL * sizeof(__bf16));

  // 1) bf16 conversions
  const int total = N * DMODEL;
  cvt_f32_bf16<<<(total + 255) / 256, 256, 0, stream>>>(x, xb, total);
  cvt_f32_bf16<<<(WW + 255) / 256, 256, 0, stream>>>(Wq, wqb, WW);
  cvt_f32_bf16<<<(WW + 255) / 256, 256, 0, stream>>>(Wk, wkb, WW);
  cvt_f32_bf16<<<(WW + 255) / 256, 256, 0, stream>>>(Wv, wvb, WW);
  cvt_f32_bf16<<<(WW + 255) / 256, 256, 0, stream>>>(Wo, wob, WW);

  // 2) Q/K/V projections via bf16 WMMA (32x256 block tile)
  gemm_xwt<<<N / 32, 256, 0, stream>>>(xb, wqb, nullptr, Qf);
  gemm_xwt<<<N / 32, 256, 0, stream>>>(xb, wkb, nullptr, Kf);
  gemm_xwt<<<N / 32, 256, 0, stream>>>(xb, wvb, nullptr, Vf);

  // 3) softmax collapse: baseline numerator = sum_j V[j], denom = N
  col_sum<<<DMODEL, 256, 0, stream>>>(Vf, vsum, N);
  init_num_denom<<<N, 256, 0, stream>>>(vsum, num, den, N);

  // 4) sparse edge corrections
  edge_scatter<<<(E * NHEAD + 7) / 8, 256, 0, stream>>>(ei, E, Qf, Kf, Vf, num, den);

  // 5) normalize + bf16, then output projection with bias via WMMA
  norm_cvt<<<N, 256, 0, stream>>>(num, den, preb);
  gemm_xwt<<<N / 32, 256, 0, stream>>>(preb, wob, bo, out);
}
